// HDGradientCompressionLayer_52664888984302
// MI455X (gfx1250) — matrix operand order repaired and checked by reading, not compiled, not run on hardware
//
#include <hip/hip_runtime.h>

// ---------------------------------------------------------------------------
// y = x @ W + b   (M=16384, N=1024, K=1024, fp32 in memory)
// Pass 1: X -> bf16 row-major in ws ; W -> bf16 col-major in ws (once each).
// Pass 2: GEMM with v_wmma_f32_16x16x32_bf16, async global->LDS staging,
//         double-buffered LDS, f32 accumulate, fused bias.
// ---------------------------------------------------------------------------

typedef __attribute__((ext_vector_type(16))) __bf16 v16bf;
typedef __attribute__((ext_vector_type(8)))  float  v8f;
typedef __attribute__((ext_vector_type(2)))  __bf16 v2bf;
typedef __attribute__((ext_vector_type(4)))  int    v4i;

#define M_TOT 16384
#define N_TOT 1024
#define K_TOT 1024
#define BM 128          // block tile rows (4 waves of 32)
#define BN 64           // block tile cols (2 waves of 32)
#define KT 32           // K per WMMA step
#define NTILES (K_TOT / KT)

union Frag16 {
    v16bf bf;
    uint4 u4[2];
};

// ---------------- bf16 conversion helpers ----------------
__device__ __forceinline__ unsigned short f32_to_bf16_rne(float f) {
    unsigned int u = __float_as_uint(f);
    unsigned int r = u + 0x7FFFu + ((u >> 16) & 1u);
    return (unsigned short)(r >> 16);
}

__device__ __forceinline__ unsigned int pack_bf16x2(float lo, float hi) {
#if __has_builtin(__builtin_amdgcn_cvt_pk_bf16_f32)
    v2bf p = __builtin_amdgcn_cvt_pk_bf16_f32(lo, hi);
    return __builtin_bit_cast(unsigned int, p);
#else
    return (unsigned int)f32_to_bf16_rne(lo) |
           ((unsigned int)f32_to_bf16_rne(hi) << 16);
#endif
}

// ---------------- async global->LDS path (gfx1250) ----------------
#if __has_builtin(__builtin_amdgcn_global_load_async_to_lds_b128)
#define HAVE_ASYNC_LDS 1
#endif

__device__ __forceinline__ void wait_async0() {
#if __has_builtin(__builtin_amdgcn_s_wait_asynccnt)
    __builtin_amdgcn_s_wait_asynccnt(0);
#else
    asm volatile("s_wait_asynccnt 0x0" ::: "memory");
#endif
}

__device__ __forceinline__ void async_copy16(const unsigned short* g,
                                             unsigned short* l) {
#ifdef HAVE_ASYNC_LDS
    // param 0: generic pointer to 16B int vector (per compiler diagnostic)
    __builtin_amdgcn_global_load_async_to_lds_b128(
        (v4i*)g, (v4i*)l, 0, 0);
#else
    *(uint4*)l = *(const uint4*)g;
#endif
}

// ===========================================================================
// Kernel 1: X (f32, [M][K]) -> Xb (bf16, [M][K]).  8 floats / thread.
// ===========================================================================
__global__ __launch_bounds__(256)
void convert_x_bf16(const float* __restrict__ X, unsigned short* __restrict__ Xb)
{
    size_t i = (size_t)blockIdx.x * 256 + threadIdx.x;   // 8 floats each
    const float4* src = (const float4*)X + i * 2;
    float4 v0 = src[0];
    float4 v1 = src[1];
    uint4 o;
    o.x = pack_bf16x2(v0.x, v0.y);
    o.y = pack_bf16x2(v0.z, v0.w);
    o.z = pack_bf16x2(v1.x, v1.y);
    o.w = pack_bf16x2(v1.z, v1.w);
    ((uint4*)Xb)[i] = o;
}

// ===========================================================================
// Kernel 2: W (f32, [K][N]) -> Wt (bf16, [N][K]) via LDS 32x32 tile transpose.
// ===========================================================================
__global__ __launch_bounds__(256)
void convert_w_bf16_t(const float* __restrict__ W, unsigned short* __restrict__ Wt)
{
    __shared__ unsigned short tile[32][33];
    const int tid = threadIdx.x;
    const int n0 = blockIdx.x * 32;
    const int k0 = blockIdx.y * 32;

    #pragma unroll
    for (int p = 0; p < 4; ++p) {
        int f  = tid + p * 256;
        int kr = f >> 5;           // 0..31
        int nc = f & 31;
        tile[nc][kr] = f32_to_bf16_rne(W[(size_t)(k0 + kr) * N_TOT + n0 + nc]);
    }
    __syncthreads();

    int nr = tid >> 3;             // 0..31
    int kc = (tid & 7) * 4;        // 0..28
    unsigned short* dst = Wt + (size_t)(n0 + nr) * K_TOT + k0 + kc;
    dst[0] = tile[nr][kc + 0];
    dst[1] = tile[nr][kc + 1];
    dst[2] = tile[nr][kc + 2];
    dst[3] = tile[nr][kc + 3];
}

// ===========================================================================
// Kernel 3: GEMM.  Xb bf16 [M][K], Wt bf16 [N][K] (col-major W).
// Double-buffered LDS, async staging, 2x2 WMMA frags per wave.
// ===========================================================================
__global__ __launch_bounds__(256)
void gemm_wmma_pre(const unsigned short* __restrict__ Xb,
                   const unsigned short* __restrict__ Wt,
                   const float* __restrict__ Bv,
                   float* __restrict__ Y)
{
    __shared__ __align__(16) unsigned short As[2][BM * KT];  // 2 x 8 KiB
    __shared__ __align__(16) unsigned short Bs[2][BN * KT];  // 2 x 4 KiB

    const int tid  = threadIdx.x;
    const int lane = tid & 31;
    const int wid  = tid >> 5;
    const int half = lane >> 4;
    const int r    = lane & 15;

    const int colBase = blockIdx.x * BN;
    const int rowBase = blockIdx.y * BM;
    const int waveM   = (wid & 3) * 32;
    const int waveN   = (wid >> 2) * 32;

    v8f acc[2][2] = {};

    auto stage = [&](int buf, int k0) {
        // A tile: 128 rows x 32 bf16 = 512 x 16B chunks, 2 per thread
        #pragma unroll
        for (int p = 0; p < 2; ++p) {
            int c   = tid + p * 256;
            int row = c >> 2;
            int off = (c & 3) * 8;                     // ushorts (16B chunks)
            async_copy16(Xb + (size_t)(rowBase + row) * K_TOT + k0 + off,
                         &As[buf][row * KT + off]);
        }
        // B tile: 64 cols x 32 bf16 = 256 x 16B chunks, 1 per thread
        {
            int col = tid >> 1;
            int off = (tid & 1) * 8;
            async_copy16(Wt + (size_t)(colBase + col) * K_TOT + k0 + off,
                         &Bs[buf][col * KT + off]);
        }
    };

    stage(0, 0);
    wait_async0();
    __syncthreads();

    for (int t = 0; t < NTILES; ++t) {
        const int buf = t & 1;
        if (t + 1 < NTILES) stage(buf ^ 1, (t + 1) * KT);

        // ---- fragments per ISA 7.12.2 16-bit layouts ----
        Frag16 a[2], b[2];
        #pragma unroll
        for (int i = 0; i < 2; ++i) {
            int row = waveM + i * 16 + r;
            a[i].u4[0] = *(const uint4*)&As[buf][row * KT + half * 8];
            a[i].u4[1] = *(const uint4*)&As[buf][row * KT + 16 + half * 8];
        }
        #pragma unroll
        for (int j = 0; j < 2; ++j) {
            int col = waveN + j * 16 + r;
            b[j].u4[0] = *(const uint4*)&Bs[buf][col * KT + half * 8];
            b[j].u4[1] = *(const uint4*)&Bs[buf][col * KT + 16 + half * 8];
        }

        #pragma unroll
        for (int i = 0; i < 2; ++i)
            #pragma unroll
            for (int j = 0; j < 2; ++j)
                acc[i][j] = __builtin_amdgcn_wmma_f32_16x16x32_bf16(
                    false, a[i].bf, false, b[j].bf,
                    (short)0, acc[i][j], false, false);

        wait_async0();
        __syncthreads();
    }

    // ---- epilogue: bias + store (C/D layout: m = v + 8*half, n = r) ----
    #pragma unroll
    for (int j = 0; j < 2; ++j) {
        int col    = colBase + waveN + j * 16 + r;
        float bias = Bv[col];
        #pragma unroll
        for (int i = 0; i < 2; ++i) {
            int rbase = rowBase + waveM + i * 16 + half * 8;
            #pragma unroll
            for (int v = 0; v < 8; ++v) {
                Y[(size_t)(rbase + v) * N_TOT + col] = acc[i][j][v] + bias;
            }
        }
    }
}

// ===========================================================================
// Fallback: single-pass kernel (round-1 style) if workspace is too small.
// ===========================================================================
__global__ __launch_bounds__(256)
void gemm_onepass(const float* __restrict__ X, const float* __restrict__ W,
                  const float* __restrict__ Bv, float* __restrict__ Y)
{
    const int AS = 40, BS = 40;
    __shared__ unsigned short Asl[BM * 40];
    __shared__ unsigned short Bsl[BN * 40];

    const int tid = threadIdx.x, lane = tid & 31, wid = tid >> 5;
    const int half = lane >> 4, r = lane & 15;
    const int colBase = blockIdx.x * BN, rowBase = blockIdx.y * BM;
    const int waveM = (wid & 3) * 32, waveN = (wid >> 2) * 32;

    v8f acc[2][2] = {};

    for (int k0 = 0; k0 < K_TOT; k0 += KT) {
        #pragma unroll
        for (int p = 0; p < 4; ++p) {
            int f = tid + p * 256, row = f >> 3, c4 = f & 7;
            const float4 v = *(const float4*)(X + (size_t)(rowBase + row) * K_TOT + k0 + c4 * 4);
            unsigned int* d = (unsigned int*)&Asl[row * AS + c4 * 4];
            d[0] = pack_bf16x2(v.x, v.y);
            d[1] = pack_bf16x2(v.z, v.w);
        }
        #pragma unroll
        for (int p = 0; p < 2; ++p) {
            int f = tid + p * 256, kr = f >> 4, c4 = f & 15;
            const float4 v = *(const float4*)(W + (size_t)(k0 + kr) * N_TOT + colBase + c4 * 4);
            Bsl[(c4 * 4 + 0) * BS + kr] = f32_to_bf16_rne(v.x);
            Bsl[(c4 * 4 + 1) * BS + kr] = f32_to_bf16_rne(v.y);
            Bsl[(c4 * 4 + 2) * BS + kr] = f32_to_bf16_rne(v.z);
            Bsl[(c4 * 4 + 3) * BS + kr] = f32_to_bf16_rne(v.w);
        }
        __syncthreads();

        Frag16 a[2], b[2];
        #pragma unroll
        for (int i = 0; i < 2; ++i) {
            int row = waveM + i * 16 + r;
            a[i].u4[0] = *(const uint4*)&Asl[row * AS + half * 8];
            a[i].u4[1] = *(const uint4*)&Asl[row * AS + 16 + half * 8];
        }
        #pragma unroll
        for (int j = 0; j < 2; ++j) {
            int col = waveN + j * 16 + r;
            b[j].u4[0] = *(const uint4*)&Bsl[col * BS + half * 8];
            b[j].u4[1] = *(const uint4*)&Bsl[col * BS + 16 + half * 8];
        }
        #pragma unroll
        for (int i = 0; i < 2; ++i)
            #pragma unroll
            for (int j = 0; j < 2; ++j)
                acc[i][j] = __builtin_amdgcn_wmma_f32_16x16x32_bf16(
                    false, a[i].bf, false, b[j].bf,
                    (short)0, acc[i][j], false, false);
        __syncthreads();
    }

    #pragma unroll
    for (int j = 0; j < 2; ++j) {
        int col = colBase + waveN + j * 16 + r;
        float bias = Bv[col];
        #pragma unroll
        for (int i = 0; i < 2; ++i) {
            int rbase = rowBase + waveM + i * 16 + half * 8;
            #pragma unroll
            for (int v = 0; v < 8; ++v)
                Y[(size_t)(rbase + v) * N_TOT + col] = acc[i][j][v] + bias;
        }
    }
}

// ===========================================================================
extern "C" void kernel_launch(void* const* d_in, const int* in_sizes, int n_in,
                              void* d_out, int out_size, void* d_ws, size_t ws_size,
                              hipStream_t stream) {
    const float* X  = (const float*)d_in[0];   // (4,4096,1024) f32
    const float* W  = (const float*)d_in[1];   // (1024,1024)   f32
    const float* Bv = (const float*)d_in[2];   // (1024,)       f32
    float*       Y  = (float*)d_out;

    const size_t xBytes = (size_t)M_TOT * K_TOT * 2;   // 32 MiB bf16
    const size_t wBytes = (size_t)N_TOT * K_TOT * 2;   //  2 MiB bf16

    if (ws_size >= xBytes + wBytes) {
        unsigned short* Xb = (unsigned short*)d_ws;
        unsigned short* Wt = (unsigned short*)((char*)d_ws + xBytes);

        // one-time per-launch conversions
        hipLaunchKernelGGL(convert_x_bf16,
                           dim3((M_TOT * (size_t)K_TOT) / (256 * 8)), dim3(256),
                           0, stream, X, Xb);
        hipLaunchKernelGGL(convert_w_bf16_t,
                           dim3(N_TOT / 32, K_TOT / 32), dim3(256),
                           0, stream, W, Wt);

        hipLaunchKernelGGL(gemm_wmma_pre,
                           dim3(N_TOT / BN, M_TOT / BM), dim3(256),
                           0, stream, Xb, Wt, Bv, Y);
    } else {
        hipLaunchKernelGGL(gemm_onepass,
                           dim3(N_TOT / BN, M_TOT / BM), dim3(256),
                           0, stream, X, W, Bv, Y);
    }
}
